// StructuredUAVObsEncoder_14078902797078
// MI455X (gfx1250) — compile-verified, matrix operand order checked by reading, not run on hardware
//
#include <hip/hip_runtime.h>
#include <hip/hip_bf16.h>

// ---------------------------------------------------------------------------
// StructuredUAVObsEncoder fused forward for MI455X (gfx1250, wave32, WMMA).
//
// Pipeline:
//   prep:  f32->bf16 weight conversion (+K padding), algebraic k/v weight
//          fusion Wkf = wk@kp_w, per-(b,a) additive bias (slot+role+team+b2s),
//          same-team scalar, key-padding keep mask.
//   main:  one workgroup per (b, 16 timesteps). All GEMMs via
//          v_wmma_f32_16x16x32_bf16 (f32 accumulate), attention/softmax/LN
//          via VALU on LDS. Output written directly from last WMMA stage.
//
// Round-2 changes:
//   * branchless exact mish epilogue (tanh(softplus(x)) = n/(n+2), n=e^2+2e)
//     -> removes libm tanhf/expf divergent slow paths between WMMA ops.
//   * global_prefetch_b8 warm-up of L2-resident bf16 weights.
// ---------------------------------------------------------------------------

typedef __attribute__((ext_vector_type(16))) __bf16 bf16x16;
typedef __attribute__((ext_vector_type(8)))  __bf16 bf16x8;
typedef __attribute__((ext_vector_type(8)))  float  f32x8;

#define TT     16          // timesteps per workgroup
#define AG     6           // agents
#define MROWS  (TT*AG)     // 96 per-agent rows
#define LDX32  40          // bf16 stride for K=32 staging (80B, 16B-mult)
#define LDH    136         // bf16 stride for 128-wide hidden (272B)
#define LDBF   272         // bf16 stride for 256-wide (544B)
#define LDF    264         // f32  stride for 256-wide (1056B)

// LDS layout (byte offsets). Two overlapping "phases"; liveness checked:
//   phase1: Xase/Xrel/H1a/H1r/MEMF live until G4; MEMB produced by G4.
//   phase2: K2/V2/QF/QBF/Q2F/... reuse the phase1 region; ATTNP/XN/HBF
//           reuse the MEMB region after MEMB is consumed.
#define OFF_XASE   0        // 96*40*2   = 7680
#define OFF_XREL   7680     // 7680
#define OFF_H1A    15360    // 96*136*2  = 26112
#define OFF_H1R    41472    // 26112
#define OFF_MEMF   67584    // 96*264*4  = 101376
#define OFF_MEMB   168960   // 96*272*2  = 52224   -> 221184
#define OFF_K2     0        // 52224
#define OFF_V2     52224    // 52224
#define OFF_QF     104448   // 16*264*4  = 16896
#define OFF_QBF    121344   // 16*272*2  = 8704
#define OFF_Q2F    130048   // 16896
#define OFF_SC     146944   // 16*24*4   = 1536
#define OFF_WSM    148480   // 1536
#define OFF_ATTNB  151040   // 8704
#define OFF_ATTNP  159744   // 16896  (overlaps dead MEMB tail)
#define OFF_XN     176640   // 8704   (dead MEMB)
#define OFF_HBF    185344   // 8704   (dead MEMB)
#define OFF_REDS   194048   // 1024
#define OFF_REDQ   195072   // 1024
#define OFF_MST    196096   // 128
#define LDS_BYTES  221184

// ---------------------------------------------------------------------------
// WMMA fragment loader. ISA 7.12.2 16-bit A layout: lane m=l&15, half g=l>>4,
// element j of v16bf maps to k = (j<8 ? g*8+j : 16+g*8+(j-8)) within a 32-wide
// K step -> two contiguous 16B runs per lane. Same pattern serves the B
// operand when W is stored row-major [n][k] (row index per lane = n).
// ---------------------------------------------------------------------------
__device__ __forceinline__ bf16x16 load_frag(const __bf16* base, int ld,
                                             int row0, int k0) {
  const int lane = threadIdx.x & 31;
  const int r = row0 + (lane & 15);
  const int g = lane >> 4;
  const __bf16* p = base + (size_t)r * ld + k0 + g * 8;
  bf16x8 lo = *(const bf16x8*)(p);
  bf16x8 hi = *(const bf16x8*)(p + 16);
  bf16x16 f;
#pragma unroll
  for (int i = 0; i < 8; ++i) { f[i] = lo[i]; f[8 + i] = hi[i]; }
  return f;
}

// Branchless mish: tanh(softplus(x)) = ((1+e)^2-1)/((1+e)^2+1) = n/(n+2),
// n = e^2+2e, e = exp(x). Clamp x<=30 so n stays finite; there n/(n+2) == 1
// to fp32 precision, so mish(x) == x exactly as required. One v_exp_f32 +
// one v_rcp_f32 + a few VALU ops, EXEC stays full (WMMA-friendly).
__device__ __forceinline__ float mish_f(float x) {
  float e = __expf(fminf(x, 30.f));
  float n = e * (e + 2.f);
  return x * n * __builtin_amdgcn_rcpf(n + 2.f);
}

// Cooperative workgroup GEMM: Y[M x N] = act(X[M x K] * W^T (+bias)(+rb6)(+acc))
// X: LDS bf16 (row stride ldx). W: global bf16, row-major [N][K].
// C/D f32 layout (ISA): lane n=l&15, g=l>>4, vgpr r -> m = 16*mt + g*8 + r.
template<bool MISH, bool ACCUM, bool ROWBIAS6>
__device__ __forceinline__ void wg_gemm(
    const __bf16* __restrict__ X, int ldx,
    const __bf16* __restrict__ W, int ldw,
    const float*  __restrict__ bias,
    const float*  __restrict__ rb6,          // per-(row%6) bias [6][256]
    const float*  __restrict__ accsrc, int ldacc,
    float*        __restrict__ Yf, int ldyf,
    __bf16*       __restrict__ Yb, int ldyb,
    int mtiles, int ntiles, int ksteps) {
  const int tid    = threadIdx.x;
  const int wave   = tid >> 5;
  const int lane   = tid & 31;
  const int nwaves = blockDim.x >> 5;
  const int total  = mtiles * ntiles;
  for (int t = wave; t < total; t += nwaves) {       // wave-uniform: EXEC full
    const int mt = t / ntiles, nt = t % ntiles;
    f32x8 acc = {};
    for (int ks = 0; ks < ksteps; ++ks) {
      bf16x16 a = load_frag(X, ldx, mt * 16, ks * 32);
      bf16x16 bfr = load_frag(W, ldw, nt * 16, ks * 32);
      acc = __builtin_amdgcn_wmma_f32_16x16x32_bf16(
          false, a, false, bfr, (short)0, acc, false, false);
    }
    const int n  = nt * 16 + (lane & 15);
    const int g  = lane >> 4;
    const float bn = bias ? bias[n] : 0.f;
#pragma unroll
    for (int r = 0; r < 8; ++r) {
      const int m = mt * 16 + g * 8 + r;
      float v = acc[r] + bn;
      if (ROWBIAS6) v += rb6[(m % AG) * 256 + n];
      if (ACCUM)    v += accsrc[(size_t)m * ldacc + n];
      if (MISH)     v = mish_f(v);
      if (Yf) Yf[(size_t)m * ldyf + n] = v;
      if (Yb) Yb[(size_t)m * ldyb + n] = (__bf16)v;
    }
  }
}

// ---------------------------------------------------------------------------
// Prep kernels
// ---------------------------------------------------------------------------
__global__ void k_cvt(const float* __restrict__ src, __bf16* __restrict__ dst,
                      int rows, int ksrc, int kdst) {
  int i = blockIdx.x * blockDim.x + threadIdx.x;
  if (i >= rows * kdst) return;
  int r = i / kdst, c = i % kdst;
  float v = (c < ksrc) ? src[r * ksrc + c] : 0.f;
  dst[i] = (__bf16)v;
}

// Wf = A @ Bm  (256x256 @ 256x256), bf = A @ bsrc + badd
__global__ void k_fuse(const float* __restrict__ A, const float* __restrict__ Bm,
                       const float* __restrict__ bsrc, const float* __restrict__ badd,
                       __bf16* __restrict__ Wf, float* __restrict__ bf) {
  int i = blockIdx.x * blockDim.x + threadIdx.x;
  if (i < 256 * 256) {
    int n = i >> 8, k = i & 255;
    float s = 0.f;
    for (int j = 0; j < 256; ++j) s += A[n * 256 + j] * Bm[j * 256 + k];
    Wf[i] = (__bf16)s;
  }
  if (i < 256) {
    float s = badd[i];
    for (int j = 0; j < 256; ++j) s += A[i * 256 + j] * bsrc[j];
    bf[i] = s;
  }
}

// base[b][a][n] = slot[a]+role_emb[rid]+team_w@team+team_b+ase_b2+rel_b2
// same[b][a] = team[b,a]·team[b,0]; keep[b][a] per reference key-padding mask.
__global__ void k_base(const float* __restrict__ team, const float* __restrict__ valid,
                       const float* __restrict__ social, const int* __restrict__ role,
                       const float* __restrict__ slot_e, const float* __restrict__ role_e,
                       const float* __restrict__ team_w, const float* __restrict__ team_b,
                       const float* __restrict__ ase_b2, const float* __restrict__ rel_b2,
                       float* __restrict__ base, float* __restrict__ same,
                       float* __restrict__ keep) {
  int i = blockIdx.x * blockDim.x + threadIdx.x;
  if (i >= 512 * AG * 256) return;
  int n = i & 255, ba = i >> 8, a = ba % AG, b = ba / AG;
  int rid = role[ba];
  float t0 = team[ba * 2 + 0], t1 = team[ba * 2 + 1];
  base[i] = slot_e[a * 256 + n] + role_e[rid * 256 + n]
          + team_w[n * 2 + 0] * t0 + team_w[n * 2 + 1] * t1
          + team_b[n] + ase_b2[n] + rel_b2[n];
  if (n == 0) {
    float s0 = team[(b * AG) * 2 + 0], s1 = team[(b * AG) * 2 + 1];
    same[ba] = t0 * s0 + t1 * s1;
    bool kp = (a == 0) || ((valid[ba] > 0.5f) && (social[b * 36 + a] > 0.5f));
    keep[ba] = kp ? 1.f : 0.f;
  }
}

// ---------------------------------------------------------------------------
// Main fused kernel
// ---------------------------------------------------------------------------
struct MainArgs {
  const float *obs, *base, *same, *keep;
  const __bf16 *ase1p, *ase2b, *rel1p, *rel2b, *sqwb, *wqb, *wkfb, *wvfb,
               *outwb, *opw1b, *opw2b;
  const float *b_ase1, *b_rel1, *b_sq, *b_q, *bkf, *bvf, *b_out, *b_op1,
              *b_op2, *ln_g, *ln_b;
  float* out;
};

__global__ __launch_bounds__(256) void uav_main(MainArgs A) {
  extern __shared__ __align__(16) char smem[];
  const int tid = threadIdx.x;
  const int b   = blockIdx.y;
  const int t0  = blockIdx.x * TT;

  __bf16* Xase  = (__bf16*)(smem + OFF_XASE);
  __bf16* Xrel  = (__bf16*)(smem + OFF_XREL);
  __bf16* H1a   = (__bf16*)(smem + OFF_H1A);
  __bf16* H1r   = (__bf16*)(smem + OFF_H1R);
  float*  MEMF  = (float*) (smem + OFF_MEMF);
  __bf16* MEMB  = (__bf16*)(smem + OFF_MEMB);
  __bf16* K2    = (__bf16*)(smem + OFF_K2);
  __bf16* V2    = (__bf16*)(smem + OFF_V2);
  float*  QF    = (float*) (smem + OFF_QF);
  __bf16* QBF   = (__bf16*)(smem + OFF_QBF);
  float*  Q2F   = (float*) (smem + OFF_Q2F);
  float*  SC    = (float*) (smem + OFF_SC);
  float*  WSM   = (float*) (smem + OFF_WSM);
  __bf16* ATTNB = (__bf16*)(smem + OFF_ATTNB);
  float*  ATTNP = (float*) (smem + OFF_ATTNP);
  __bf16* XN    = (__bf16*)(smem + OFF_XN);
  __bf16* HBF   = (__bf16*)(smem + OFF_HBF);
  float*  REDS  = (float*) (smem + OFF_REDS);
  float*  REDQ  = (float*) (smem + OFF_REDQ);
  float*  MST   = (float*) (smem + OFF_MST);

  // Warm L2/near caches for the shared bf16 weights (global_prefetch_b8).
  // 4 weight arrays, prefetch one 256B line per thread per array slice.
  {
    const char* wp[4] = {(const char*)A.wkfb, (const char*)A.wvfb,
                         (const char*)A.sqwb, (const char*)A.opw1b};
#pragma unroll
    for (int w = 0; w < 4; ++w)
      __builtin_prefetch(wp[w] + (size_t)tid * 512, 0, 1);
  }

  const float* obsB = A.obs + (((size_t)b * 64 + t0) * AG) * 7;  // [16][6][7]

  // ---- stage inputs (zero-padded K=32) ----
  for (int i = tid; i < MROWS * LDX32; i += 256) {
    int r = i / LDX32, c = i % LDX32;
    Xase[i] = (__bf16)((c < 7) ? obsB[r * 7 + c] : 0.f);
  }
  for (int i = tid; i < MROWS * LDX32; i += 256) {
    int r = i / LDX32, c = i % LDX32;
    int t = r / AG, a = r % AG;
    float v = 0.f;
    if (c < 7)       v = obsB[r * 7 + c] - obsB[(t * AG) * 7 + c];
    else if (c == 7) v = A.same[b * AG + a];
    Xrel[i] = (__bf16)v;
  }
  __syncthreads();

  // ---- agent + relation MLPs ----
  wg_gemm<true ,false,false>(Xase, LDX32, A.ase1p, 32, A.b_ase1, nullptr,
                             nullptr, 0, nullptr, 0, H1a, LDH, 6, 8, 1);
  wg_gemm<true ,false,false>(Xrel, LDX32, A.rel1p, 32, A.b_rel1, nullptr,
                             nullptr, 0, nullptr, 0, H1r, LDH, 6, 8, 1);
  __syncthreads();
  wg_gemm<false,false,false>(H1a, LDH, A.ase2b, 128, nullptr, nullptr,
                             nullptr, 0, MEMF, LDF, nullptr, 0, 6, 16, 4);
  __syncthreads();
  // memory = ase2 + rel2 + base(b,a)   -> bf16 MEMB
  wg_gemm<false,true ,true >(H1r, LDH, A.rel2b, 128, nullptr,
                             A.base + (size_t)b * AG * 256, MEMF, LDF,
                             nullptr, 0, MEMB, LDBF, 6, 16, 4);
  __syncthreads();

  // ---- fused k2/v2 projections + q ----
  wg_gemm<false,false,false>(MEMB, LDBF, A.wkfb, 256, A.bkf, nullptr,
                             nullptr, 0, nullptr, 0, K2, LDBF, 6, 16, 8);
  wg_gemm<false,false,false>(MEMB, LDBF, A.wvfb, 256, A.bvf, nullptr,
                             nullptr, 0, nullptr, 0, V2, LDBF, 6, 16, 8);
  // q over a==0 rows: row stride trick ldx = 6*LDBF selects rows t*6
  wg_gemm<false,false,false>(MEMB, AG * LDBF, A.sqwb, 256, A.b_sq, nullptr,
                             nullptr, 0, QF, LDF, QBF, LDBF, 1, 16, 8);
  __syncthreads();
  wg_gemm<false,false,false>(QBF, LDBF, A.wqb, 256, A.b_q, nullptr,
                             nullptr, 0, Q2F, LDF, nullptr, 0, 1, 16, 8);
  __syncthreads();

  // ---- attention: scores (masked), softmax over A=6, weighted V ----
  for (int idx = tid; idx < TT * 4 * AG; idx += 256) {
    int t = idx / 24, rem = idx % 24, h = rem / AG, a = rem % AG;
    const float*  qp = Q2F + (size_t)t * LDF + h * 64;
    const __bf16* kp = K2 + ((size_t)t * AG + a) * LDBF + h * 64;
    float s = 0.f;
    for (int d = 0; d < 64; ++d) s += qp[d] * (float)kp[d];
    s *= 0.125f;                                  // 1/sqrt(64)
    if (A.keep[b * AG + a] < 0.5f) s = -1e9f;
    SC[idx] = s;
  }
  __syncthreads();
  if (tid < TT * 4) {
    int t = tid >> 2, h = tid & 3;
    float mx = -3.4e38f, e[AG];
    for (int a = 0; a < AG; ++a) mx = fmaxf(mx, SC[t * 24 + h * AG + a]);
    float ssum = 0.f;
    for (int a = 0; a < AG; ++a) {
      e[a] = __expf(SC[t * 24 + h * AG + a] - mx); ssum += e[a];
    }
    float inv = __builtin_amdgcn_rcpf(ssum);
    for (int a = 0; a < AG; ++a) WSM[t * 24 + h * AG + a] = e[a] * inv;
  }
  __syncthreads();
  for (int i = tid; i < TT * 256; i += 256) {
    int t = i >> 8, c = i & 255, h = c >> 6;
    float s = 0.f;
    for (int a = 0; a < AG; ++a)
      s += WSM[t * 24 + h * AG + a] * (float)V2[((size_t)t * AG + a) * LDBF + c];
    ATTNB[(size_t)t * LDBF + c] = (__bf16)s;
  }
  __syncthreads();
  wg_gemm<false,false,false>(ATTNB, LDBF, A.outwb, 256, A.b_out, nullptr,
                             nullptr, 0, ATTNP, LDF, nullptr, 0, 1, 16, 8);
  __syncthreads();

  // ---- x = q + attn; LayerNorm (biased var, eps 1e-5) ----
  {
    int row = tid >> 4, part = tid & 15;
    float s = 0.f, q = 0.f;
    for (int i2 = 0; i2 < 16; ++i2) {
      int c = part * 16 + i2;
      float xv = QF[(size_t)row * LDF + c] + ATTNP[(size_t)row * LDF + c];
      s += xv; q += xv * xv;
    }
    REDS[tid] = s; REDQ[tid] = q;
  }
  __syncthreads();
  if (tid < TT) {
    float s = 0.f, q = 0.f;
    for (int i2 = 0; i2 < 16; ++i2) { s += REDS[tid * 16 + i2]; q += REDQ[tid * 16 + i2]; }
    float mu  = s * (1.f / 256.f);
    float var = q * (1.f / 256.f) - mu * mu;
    MST[tid * 2] = mu; MST[tid * 2 + 1] = rsqrtf(var + 1e-5f);
  }
  __syncthreads();
  for (int i = tid; i < TT * 256; i += 256) {
    int r = i >> 8, c = i & 255;
    float xv = QF[(size_t)r * LDF + c] + ATTNP[(size_t)r * LDF + c];
    float xn = (xv - MST[r * 2]) * MST[r * 2 + 1] * A.ln_g[c] + A.ln_b[c];
    XN[(size_t)r * LDBF + c] = (__bf16)xn;
  }
  __syncthreads();

  // ---- output MLP: Linear -> Mish -> Linear (last stage writes global) ----
  wg_gemm<true ,false,false>(XN, LDBF, A.opw1b, 256, A.b_op1, nullptr,
                             nullptr, 0, nullptr, 0, HBF, LDBF, 1, 16, 8);
  __syncthreads();
  float* outg = A.out + ((size_t)b * 64 + t0) * 256;
  wg_gemm<false,false,false>(HBF, LDBF, A.opw2b, 256, A.b_op2, nullptr,
                             nullptr, 0, outg, 256, nullptr, 0, 1, 16, 8);
}

// ---------------------------------------------------------------------------
// Host launcher. Inputs in setup_inputs() insertion order with params dict
// flattened in _make_params() insertion order.
// ---------------------------------------------------------------------------
extern "C" void kernel_launch(void* const* d_in, const int* in_sizes, int n_in,
                              void* d_out, int out_size, void* d_ws, size_t ws_size,
                              hipStream_t stream) {
  (void)in_sizes; (void)n_in; (void)out_size; (void)ws_size;
  const float* obs    = (const float*)d_in[0];
  const float* team   = (const float*)d_in[1];
  const float* valid  = (const float*)d_in[2];
  const float* social = (const float*)d_in[3];
  const int*   role   = (const int*)  d_in[4];
  const float* ase_w1 = (const float*)d_in[5];
  const float* ase_b1 = (const float*)d_in[6];
  const float* ase_w2 = (const float*)d_in[7];
  const float* ase_b2 = (const float*)d_in[8];
  const float* team_w = (const float*)d_in[9];
  const float* team_b = (const float*)d_in[10];
  const float* slot_e = (const float*)d_in[11];
  const float* role_e = (const float*)d_in[12];
  const float* rel_w1 = (const float*)d_in[13];
  const float* rel_b1 = (const float*)d_in[14];
  const float* rel_w2 = (const float*)d_in[15];
  const float* rel_b2 = (const float*)d_in[16];
  const float* sq_w   = (const float*)d_in[17];
  const float* sq_b   = (const float*)d_in[18];
  const float* kp_w   = (const float*)d_in[19];
  const float* kp_b   = (const float*)d_in[20];
  const float* vp_w   = (const float*)d_in[21];
  const float* vp_b   = (const float*)d_in[22];
  const float* in_w   = (const float*)d_in[23];
  const float* in_b   = (const float*)d_in[24];
  const float* out_w  = (const float*)d_in[25];
  const float* out_b  = (const float*)d_in[26];
  const float* ln_g   = (const float*)d_in[27];
  const float* ln_b   = (const float*)d_in[28];
  const float* op_w1  = (const float*)d_in[29];
  const float* op_b1  = (const float*)d_in[30];
  const float* op_w2  = (const float*)d_in[31];
  const float* op_b2  = (const float*)d_in[32];

  char* ws = (char*)d_ws;
  __bf16* ase1p = (__bf16*)(ws + 0);
  __bf16* rel1p = (__bf16*)(ws + 8192);
  __bf16* ase2b = (__bf16*)(ws + 16384);
  __bf16* rel2b = (__bf16*)(ws + 81920);
  __bf16* sqwb  = (__bf16*)(ws + 147456);
  __bf16* wqb   = (__bf16*)(ws + 278528);
  __bf16* wkfb  = (__bf16*)(ws + 409600);
  __bf16* wvfb  = (__bf16*)(ws + 540672);
  __bf16* outwb = (__bf16*)(ws + 671744);
  __bf16* opw1b = (__bf16*)(ws + 802816);
  __bf16* opw2b = (__bf16*)(ws + 933888);
  float*  bkf   = (float*) (ws + 1064960);
  float*  bvf   = (float*) (ws + 1065984);
  float*  baseb = (float*) (ws + 1067008);
  float*  sameb = (float*) (ws + 4212736);
  float*  keepb = (float*) (ws + 4225024);

  // weight conversion / padding
  k_cvt<<<16,  256, 0, stream>>>(ase_w1, ase1p, 128, 7, 32);
  k_cvt<<<16,  256, 0, stream>>>(rel_w1, rel1p, 128, 8, 32);
  k_cvt<<<128, 256, 0, stream>>>(ase_w2, ase2b, 256, 128, 128);
  k_cvt<<<128, 256, 0, stream>>>(rel_w2, rel2b, 256, 128, 128);
  k_cvt<<<256, 256, 0, stream>>>(sq_w,   sqwb,  256, 256, 256);
  k_cvt<<<256, 256, 0, stream>>>(in_w,   wqb,   256, 256, 256);   // wq rows
  k_cvt<<<256, 256, 0, stream>>>(out_w,  outwb, 256, 256, 256);
  k_cvt<<<256, 256, 0, stream>>>(op_w1,  opw1b, 256, 256, 256);
  k_cvt<<<256, 256, 0, stream>>>(op_w2,  opw2b, 256, 256, 256);
  // algebraic fusion: Wkf = wk@kp_w, bkf = wk@kp_b + bk  (same for v)
  k_fuse<<<256, 256, 0, stream>>>(in_w + 256*256,   kp_w, kp_b, in_b + 256, wkfb, bkf);
  k_fuse<<<256, 256, 0, stream>>>(in_w + 2*256*256, vp_w, vp_b, in_b + 512, wvfb, bvf);
  // per-(b,a) bias, same-team scalar, keep mask
  k_base<<<3072, 256, 0, stream>>>(team, valid, social, role, slot_e, role_e,
                                   team_w, team_b, ase_b2, rel_b2,
                                   baseb, sameb, keepb);

  MainArgs args;
  args.obs = obs; args.base = baseb; args.same = sameb; args.keep = keepb;
  args.ase1p = ase1p; args.ase2b = ase2b; args.rel1p = rel1p; args.rel2b = rel2b;
  args.sqwb = sqwb; args.wqb = wqb; args.wkfb = wkfb; args.wvfb = wvfb;
  args.outwb = outwb; args.opw1b = opw1b; args.opw2b = opw2b;
  args.b_ase1 = ase_b1; args.b_rel1 = rel_b1; args.b_sq = sq_b; args.b_q = in_b;
  args.bkf = bkf; args.bvf = bvf; args.b_out = out_b;
  args.b_op1 = op_b1; args.b_op2 = op_b2; args.ln_g = ln_g; args.ln_b = ln_b;
  args.out = (float*)d_out;

  dim3 grid(64 / TT, 512);
  uav_main<<<grid, 256, LDS_BYTES, stream>>>(args);
}
